// _BaseDiT1d_8040178778692
// MI455X (gfx1250) — matrix operand order repaired and checked by reading, not compiled, not run on hardware
//
#include <hip/hip_runtime.h>
#include <math.h>
#include <stdint.h>

// ---------------------------------------------------------------------------
// CDNA5 (gfx1250) DiT block: all GEMMs via V_WMMA_F32_16X16X32_BF16, wave32.
// A-tile staging uses GLOBAL_LOAD_ASYNC_TO_LDS_B128 (ASYNCcnt) per ISA ch.10.
// ---------------------------------------------------------------------------

typedef __bf16 bf16;
typedef __attribute__((ext_vector_type(16))) __bf16 v16bf;
typedef __attribute__((ext_vector_type(8)))  float  v8f;
typedef __attribute__((ext_vector_type(4)))  unsigned int u32x4;

#define BM 128
#define BN 128
#define BK 32
#define LDK 40   // padded K-stride (halfs) in LDS; 80B rows keep 16B alignment

struct GemmArgs {
  const bf16* A; const bf16* B;
  float* C; bf16* Cb;
  const float* bias; const float* res;
  int M, N, K;
  int lda, ldb, ldc;
  int H;                                   // heads: batch z -> (b = z/H, h = z%H)
  long long aSB, aSH, bSB, bSH, cSB, cSH;  // per-batch / per-head element strides
  int flags;                               // 1=bias 2=gelu 4=residual 8=bf16 out
};

union FragAB { v16bf v; u32x4 q[2]; };

__global__ __launch_bounds__(256) void gemm_bf16_wmma(GemmArgs g) {
  __shared__ bf16 sA[2][BM * LDK];
  __shared__ bf16 sB[2][BN * LDK];

  const int tid  = threadIdx.x;
  const int lane = tid & 31;
  const int wave = tid >> 5;
  const int wm   = wave & 3;    // 4 waves along M (32 rows each)
  const int wn   = wave >> 2;   // 2 waves along N (64 cols each)
  const int lm   = lane & 15;
  const int hi   = lane >> 4;   // half-wave select

  const int mBase = blockIdx.y * BM;
  const int nBase = blockIdx.x * BN;
  const int b = (int)blockIdx.z / g.H;
  const int h = (int)blockIdx.z % g.H;

  const bf16* Ap = g.A + (long long)b * g.aSB + (long long)h * g.aSH;
  const bf16* Bp = g.B + (long long)b * g.bSB + (long long)h * g.bSH;
  const long long cOff = (long long)b * g.cSB + (long long)h * g.cSH;

  v8f acc[2][4];
#pragma unroll
  for (int i = 0; i < 2; ++i)
#pragma unroll
    for (int j = 0; j < 4; ++j)
#pragma unroll
      for (int r = 0; r < 8; ++r) acc[i][j][r] = 0.f;

  const int KT = g.K / BK;   // all K here are multiples of 32

  auto stage = [&](int buf, int k0) {
    // ---- A tile: 128 rows x 32 halfs. Each thread owns a 32B slice and
    // copies it global->LDS with two async b128 ops (same IOFFSET applies to
    // both the LDS and global side per the CDNA5 async-load definition).
    {
      int r  = tid >> 1;
      int kc = (tid & 1) << 4;
      int gr = mBase + r;
      if (gr >= g.M) gr = g.M - 1;   // clamp; OOB rows are never stored
      const bf16* gp = Ap + (long long)gr * g.lda + k0 + kc;
      unsigned lds = (unsigned)(uintptr_t)(const void*)&sA[buf][r * LDK + kc];
      unsigned long long ga = (unsigned long long)(uintptr_t)gp;
      asm volatile("global_load_async_to_lds_b128 %0, %1, off"
                   :: "v"(lds), "v"(ga) : "memory");
      asm volatile("global_load_async_to_lds_b128 %0, %1, off offset:16"
                   :: "v"(lds), "v"(ga) : "memory");
    }
    // ---- B tile: 32 x 128, stored transposed (n-major) so WMMA fragments are
    // K-contiguous. Coalesced across lanes at each kk; pointer induction keeps
    // the address math to one add per element.
    {
      int n  = tid & 127;
      int kh = (tid >> 7) << 4;       // 0 or 16
      int gn = nBase + n;
      union { bf16 e[16]; u32x4 q[2]; } tmp;
      if (gn < g.N) {
        const bf16* p = Bp + (long long)(k0 + kh) * g.ldb + gn;
#pragma unroll
        for (int kk = 0; kk < 16; ++kk) { tmp.e[kk] = *p; p += g.ldb; }
      } else {
#pragma unroll
        for (int kk = 0; kk < 16; ++kk) tmp.e[kk] = (bf16)0.f;
      }
      *(u32x4*)&sB[buf][n * LDK + kh]     = tmp.q[0];
      *(u32x4*)&sB[buf][n * LDK + kh + 8] = tmp.q[1];
    }
  };

  stage(0, 0);
  asm volatile("s_wait_asynccnt 0" ::: "memory");
  __syncthreads();

  for (int kt = 0; kt < KT; ++kt) {
    const int cur = kt & 1;
    if (kt + 1 < KT) stage(cur ^ 1, (kt + 1) * BK);

    FragAB aF[2], bF[4];
    // A frag per ISA: lane<16 -> K 0..7 & 16..23 ; lane>=16 -> K 8..15 & 24..31
#pragma unroll
    for (int i = 0; i < 2; ++i) {
      int m = wm * 32 + i * 16 + lm;
      aF[i].q[0] = *(const u32x4*)&sA[cur][m * LDK + hi * 8];
      aF[i].q[1] = *(const u32x4*)&sA[cur][m * LDK + 16 + hi * 8];
    }
    // B frag: lane n = lm, K 0..7 & 16..23 (hi=0) or 8..15 & 24..31 (hi=1)
#pragma unroll
    for (int j = 0; j < 4; ++j) {
      int n = wn * 64 + j * 16 + lm;
      bF[j].q[0] = *(const u32x4*)&sB[cur][n * LDK + hi * 8];
      bF[j].q[1] = *(const u32x4*)&sB[cur][n * LDK + 16 + hi * 8];
    }
#pragma unroll
    for (int i = 0; i < 2; ++i)
#pragma unroll
      for (int j = 0; j < 4; ++j)
        acc[i][j] = __builtin_amdgcn_wmma_f32_16x16x32_bf16(
            false, aF[i].v, false, bF[j].v, (short)0, acc[i][j], false, false);

    asm volatile("s_wait_asynccnt 0" ::: "memory");
    __syncthreads();
  }

  // Epilogue: C layout = VGPR r -> M = r + hi*8, N = lm
#pragma unroll
  for (int i = 0; i < 2; ++i) {
#pragma unroll
    for (int j = 0; j < 4; ++j) {
      int gcol = nBase + wn * 64 + j * 16 + lm;
      if (gcol >= g.N) continue;
#pragma unroll
      for (int r = 0; r < 8; ++r) {
        int grow = mBase + wm * 32 + i * 16 + hi * 8 + r;
        if (grow >= g.M) continue;
        float v = acc[i][j][r];
        if (g.flags & 1) v += g.bias[gcol];
        if (g.flags & 2) v = 0.5f * v * (1.f + erff(v * 0.70710678118654752f));
        long long idx = cOff + (long long)grow * g.ldc + gcol;
        if (g.flags & 4) v += g.res[idx];
        if (g.flags & 8) g.Cb[idx] = (bf16)v;
        else             g.C[idx]  = v;
      }
    }
  }
}

// ---------------------------------------------------------------------------
// Elementwise / reduction helper kernels (fp32 glue, bf16 outputs for GEMM A).
// ---------------------------------------------------------------------------

__global__ void f32_to_bf16_kernel(const float* __restrict__ in, bf16* __restrict__ out,
                                   long long n) {
  long long i = (long long)blockIdx.x * blockDim.x + threadIdx.x;
  long long stride = (long long)gridDim.x * blockDim.x;
  for (; i < n; i += stride) out[i] = (bf16)in[i];
}

// out[b, col] = sum_k cond[b,k] * W[k,col] + bias[col]   (tiny: B=2 rows)
__global__ void gemv_bias_kernel(const float* __restrict__ c, const float* __restrict__ W,
                                 const float* __restrict__ bias, float* __restrict__ out,
                                 int K, int Ncols) {
  int col = blockIdx.x * blockDim.x + threadIdx.x;
  int b = blockIdx.y;
  if (col >= Ncols) return;
  float acc = bias[col];
  for (int k = 0; k < K; ++k) acc += c[(long long)b * K + k] * W[(long long)k * Ncols + col];
  out[(long long)b * Ncols + col] = acc;
}

// LayerNorm (+ optional adaLN modulation), bf16 output. One block per row.
__global__ __launch_bounds__(256) void adaln_kernel(
    const float* __restrict__ x, const float* __restrict__ lw, const float* __restrict__ lb,
    const float* __restrict__ g, const float* __restrict__ bt, bf16* __restrict__ out,
    int D, int rowsPerBatch) {
  int row = blockIdx.x;
  int b = row / rowsPerBatch;
  const float* xr = x + (long long)row * D;
  __shared__ float r1[256], r2[256];
  float s = 0.f, s2 = 0.f;
  for (int d = threadIdx.x; d < D; d += 256) { float v = xr[d]; s += v; s2 += v * v; }
  r1[threadIdx.x] = s; r2[threadIdx.x] = s2;
  __syncthreads();
  for (int off = 128; off > 0; off >>= 1) {
    if (threadIdx.x < off) { r1[threadIdx.x] += r1[threadIdx.x + off];
                             r2[threadIdx.x] += r2[threadIdx.x + off]; }
    __syncthreads();
  }
  float mu  = r1[0] / (float)D;
  float var = r2[0] / (float)D - mu * mu;
  float inv = rsqrtf(var + 1e-5f);
  bf16* orow = out + (long long)row * D;
  for (int d = threadIdx.x; d < D; d += 256) {
    float v = (xr[d] - mu) * inv * lw[d] + lb[d];
    if (g) v = v * (1.f + g[(long long)b * D + d]) + bt[(long long)b * D + d];
    orow[d] = (bf16)v;
  }
}

// RoPE on Q (rows = B*N, cols = MID=1024, DH=64), fp32 in -> bf16 out.
__global__ void rope_q_kernel(const float* __restrict__ q, bf16* __restrict__ out,
                              long long totalPairs, int seqN) {
  long long idx = (long long)blockIdx.x * blockDim.x + threadIdx.x;
  if (idx >= totalPairs) return;
  int row = (int)(idx >> 9);           // 512 pairs per row
  int p   = (int)(idx & 511);
  int i   = p & 31;                    // pair index within head (DH/2 = 32)
  int n   = row % seqN;                // sequence position
  float inv = powf(10000.f, -(float)(2 * i) / 64.f);
  float sn, cs; sincosf((float)n * inv, &sn, &cs);
  long long base = (long long)row * 1024 + 2 * p;
  float xe = q[base], xo = q[base + 1];
  out[base]     = (bf16)(xe * cs - xo * sn);
  out[base + 1] = (bf16)(xe * sn + xo * cs);
}

// Split KV: K -> RoPE + transpose into kT[b][h][d][m] (bf16); V -> bf16 (row-major).
__global__ void kv_proc_kernel(const float* __restrict__ kv, bf16* __restrict__ kT,
                               bf16* __restrict__ v, int S, long long totalPairs) {
  long long idx = (long long)blockIdx.x * blockDim.x + threadIdx.x;
  if (idx >= totalPairs) return;
  int row = (int)(idx >> 9);           // 512 k-pairs per row
  int p   = (int)(idx & 511);
  int b   = row / S, m = row % S;
  int h   = p >> 5, i = p & 31;
  long long kvBase = (long long)row * 2048;
  float xe = kv[kvBase + 2 * p], xo = kv[kvBase + 2 * p + 1];
  float inv = powf(10000.f, -(float)(2 * i) / 64.f);
  float sn, cs; sincosf((float)m * inv, &sn, &cs);
  long long tb = ((long long)(b * 16 + h) * 64) * S;
  kT[tb + (long long)(2 * i) * S + m]     = (bf16)(xe * cs - xo * sn);
  kT[tb + (long long)(2 * i + 1) * S + m] = (bf16)(xe * sn + xo * cs);
  long long vBase = (long long)row * 1024 + 2 * p;
  v[vBase]     = (bf16)kv[kvBase + 1024 + 2 * p];
  v[vBase + 1] = (bf16)kv[kvBase + 1024 + 2 * p + 1];
}

// Row softmax (scaled), fp32 in -> bf16 probs. One block per row.
__global__ __launch_bounds__(256) void softmax_kernel(
    const float* __restrict__ sim, bf16* __restrict__ probs, int W, float scale) {
  int row = blockIdx.x;
  const float* sr = sim + (long long)row * W;
  __shared__ float red[256];
  float mx = -3.0e38f;
  for (int d = threadIdx.x; d < W; d += 256) mx = fmaxf(mx, sr[d] * scale);
  red[threadIdx.x] = mx; __syncthreads();
  for (int off = 128; off > 0; off >>= 1) {
    if (threadIdx.x < off) red[threadIdx.x] = fmaxf(red[threadIdx.x], red[threadIdx.x + off]);
    __syncthreads();
  }
  float MX = red[0];
  __syncthreads();
  float sum = 0.f;
  for (int d = threadIdx.x; d < W; d += 256) sum += __expf(sr[d] * scale - MX);
  red[threadIdx.x] = sum; __syncthreads();
  for (int off = 128; off > 0; off >>= 1) {
    if (threadIdx.x < off) red[threadIdx.x] += red[threadIdx.x + off];
    __syncthreads();
  }
  float rinv = 1.f / red[0];
  bf16* pr = probs + (long long)row * W;
  for (int d = threadIdx.x; d < W; d += 256)
    pr[d] = (bf16)(__expf(sr[d] * scale - MX) * rinv);
}

// ---------------------------------------------------------------------------
// Host side
// ---------------------------------------------------------------------------

static void gemm(hipStream_t s, const bf16* A, int lda, const bf16* B, int ldb,
                 float* C, bf16* Cb, int ldc, int M, int N, int K,
                 const float* bias, const float* res, int flags,
                 int batches, int H,
                 long long aSB, long long aSH, long long bSB, long long bSH,
                 long long cSB, long long cSH) {
  GemmArgs g;
  g.A = A; g.B = B; g.C = C; g.Cb = Cb; g.bias = bias; g.res = res;
  g.M = M; g.N = N; g.K = K; g.lda = lda; g.ldb = ldb; g.ldc = ldc; g.H = H;
  g.aSB = aSB; g.aSH = aSH; g.bSB = bSB; g.bSH = bSH; g.cSB = cSB; g.cSH = cSH;
  g.flags = flags;
  dim3 grid((N + BN - 1) / BN, (M + BM - 1) / BM, batches);
  gemm_bf16_wmma<<<grid, dim3(256), 0, s>>>(g);
}

static void cvt(hipStream_t s, const float* in, bf16* out, long long n) {
  int blocks = (int)((n + 255) / 256); if (blocks > 4096) blocks = 4096;
  f32_to_bf16_kernel<<<blocks, 256, 0, s>>>(in, out, n);
}

extern "C" void kernel_launch(void* const* d_in, const int* in_sizes, int n_in,
                              void* d_out, int out_size, void* d_ws, size_t ws_size,
                              hipStream_t stream) {
  (void)in_sizes; (void)n_in; (void)out_size; (void)ws_size;

  const float* x_in    = (const float*)d_in[0];
  const float* cond    = (const float*)d_in[1];
  const float* context = (const float*)d_in[2];
  const float* ln1_w = (const float*)d_in[3];  const float* ln1_b = (const float*)d_in[4];
  const float* g1_w  = (const float*)d_in[5];  const float* g1_bi = (const float*)d_in[6];
  const float* b1_w  = (const float*)d_in[7];  const float* b1_bi = (const float*)d_in[8];
  const float* ln2_w = (const float*)d_in[9];  const float* ln2_b = (const float*)d_in[10];
  const float* g2_w  = (const float*)d_in[11]; const float* g2_bi = (const float*)d_in[12];
  const float* b2_w  = (const float*)d_in[13]; const float* b2_bi = (const float*)d_in[14];
  const float* sa_wq = (const float*)d_in[15]; const float* sa_wkv = (const float*)d_in[16];
  const float* sa_wo = (const float*)d_in[17]; const float* sa_bo  = (const float*)d_in[18];
  const float* ca_lnc_w = (const float*)d_in[19]; const float* ca_lnc_b = (const float*)d_in[20];
  const float* ca_wq = (const float*)d_in[21]; const float* ca_wkv = (const float*)d_in[22];
  const float* ca_wo = (const float*)d_in[23]; const float* ca_bo  = (const float*)d_in[24];
  const float* ff_w1 = (const float*)d_in[25]; const float* ff_b1  = (const float*)d_in[26];
  const float* ff_w2 = (const float*)d_in[27]; const float* ff_b2  = (const float*)d_in[28];

  // Dimensions
  const int Bv = 2, Nv = 1024, Mv = 512, Dv = 1024, Hh = 16, MIDv = 1024, DCv = 768, FFv = 4096;
  const int rowsX = Bv * Nv;      // 2048
  const int rowsC = Bv * Mv;      // 1024

  // Workspace bump allocator (total ~300 MB)
  char* ws = (char*)d_ws;
  size_t off = 0;
  auto alloc = [&](size_t bytes) -> char* {
    char* p = ws + off; off += (bytes + 255) & ~(size_t)255; return p;
  };

  bf16* wq_sa_b  = (bf16*)alloc((size_t)Dv * MIDv * 2);
  bf16* wkv_sa_b = (bf16*)alloc((size_t)Dv * 2 * MIDv * 2);
  bf16* wo_sa_b  = (bf16*)alloc((size_t)MIDv * Dv * 2);
  bf16* wq_ca_b  = (bf16*)alloc((size_t)Dv * MIDv * 2);
  bf16* wkv_ca_b = (bf16*)alloc((size_t)DCv * 2 * MIDv * 2);
  bf16* wo_ca_b  = (bf16*)alloc((size_t)MIDv * Dv * 2);
  bf16* ff1_b    = (bf16*)alloc((size_t)Dv * FFv * 2);
  bf16* ff2_b    = (bf16*)alloc((size_t)FFv * Dv * 2);
  bf16* act_b    = (bf16*)alloc((size_t)rowsX * Dv * 2);
  bf16* ctx_b    = (bf16*)alloc((size_t)rowsC * DCv * 2);
  float* q_f     = (float*)alloc((size_t)rowsX * MIDv * 4);
  bf16* q_b      = (bf16*)alloc((size_t)rowsX * MIDv * 2);
  float* kv_f    = (float*)alloc((size_t)rowsX * 2 * MIDv * 4);
  bf16* kT_b     = (bf16*)alloc((size_t)Bv * Hh * 64 * Nv * 2);
  bf16* v_b      = (bf16*)alloc((size_t)rowsX * MIDv * 2);
  float* sim_f   = (float*)alloc((size_t)Bv * Hh * Nv * Nv * 4);
  bf16* probs_b  = (bf16*)alloc((size_t)Bv * Hh * Nv * Nv * 2);
  bf16* o_b      = (bf16*)alloc((size_t)rowsX * MIDv * 2);
  float* x1      = (float*)alloc((size_t)rowsX * Dv * 4);
  float* x2      = (float*)alloc((size_t)rowsX * Dv * 4);
  bf16* h_b      = (bf16*)alloc((size_t)rowsX * FFv * 2);
  float* mod     = (float*)alloc((size_t)4 * Bv * Dv * 4);
  float* g1 = mod;            float* b1 = mod + 2 * Dv;
  float* g2 = mod + 4 * Dv;   float* b2 = mod + 6 * Dv;

  // Weight conversions fp32 -> bf16
  cvt(stream, sa_wq,  wq_sa_b,  (long long)Dv * MIDv);
  cvt(stream, sa_wkv, wkv_sa_b, (long long)Dv * 2 * MIDv);
  cvt(stream, sa_wo,  wo_sa_b,  (long long)MIDv * Dv);
  cvt(stream, ca_wq,  wq_ca_b,  (long long)Dv * MIDv);
  cvt(stream, ca_wkv, wkv_ca_b, (long long)DCv * 2 * MIDv);
  cvt(stream, ca_wo,  wo_ca_b,  (long long)MIDv * Dv);
  cvt(stream, ff_w1,  ff1_b,    (long long)Dv * FFv);
  cvt(stream, ff_w2,  ff2_b,    (long long)FFv * Dv);

  // adaLN modulation vectors (tiny GEMVs)
  dim3 gvg((Dv + 255) / 256, Bv);
  gemv_bias_kernel<<<gvg, 256, 0, stream>>>(cond, g1_w, g1_bi, g1, Dv, Dv);
  gemv_bias_kernel<<<gvg, 256, 0, stream>>>(cond, b1_w, b1_bi, b1, Dv, Dv);
  gemv_bias_kernel<<<gvg, 256, 0, stream>>>(cond, g2_w, g2_bi, g2, Dv, Dv);
  gemv_bias_kernel<<<gvg, 256, 0, stream>>>(cond, b2_w, b2_bi, b2, Dv, Dv);

  const float iscale = 0.125f;  // DH^-0.5

  // ===================== Self attention =====================
  adaln_kernel<<<rowsX, 256, 0, stream>>>(x_in, ln1_w, ln1_b, g1, b1, act_b, Dv, Nv);
  gemm(stream, act_b, Dv, wq_sa_b, MIDv, q_f, nullptr, MIDv, rowsX, MIDv, Dv,
       nullptr, nullptr, 0, 1, 1, 0, 0, 0, 0, 0, 0);
  gemm(stream, act_b, Dv, wkv_sa_b, 2 * MIDv, kv_f, nullptr, 2 * MIDv, rowsX, 2 * MIDv, Dv,
       nullptr, nullptr, 0, 1, 1, 0, 0, 0, 0, 0, 0);
  rope_q_kernel<<<(rowsX * 512 + 255) / 256, 256, 0, stream>>>(q_f, q_b, (long long)rowsX * 512, Nv);
  kv_proc_kernel<<<(rowsX * 512 + 255) / 256, 256, 0, stream>>>(kv_f, kT_b, v_b, Nv, (long long)rowsX * 512);
  // sim[b,h] = Q[b,:,h,:] @ Kt[b,h]   (M=1024, N=1024, K=64, batch=32)
  gemm(stream, q_b, MIDv, kT_b, Nv, sim_f, nullptr, Nv, Nv, Nv, 64,
       nullptr, nullptr, 0, Bv * Hh, Hh,
       (long long)Nv * MIDv, 64, (long long)Hh * 64 * Nv, (long long)64 * Nv,
       (long long)Hh * Nv * Nv, (long long)Nv * Nv);
  softmax_kernel<<<Bv * Hh * Nv, 256, 0, stream>>>(sim_f, probs_b, Nv, iscale);
  // O = probs @ V   (M=1024, N=64, K=1024, batch=32) -> bf16 out
  gemm(stream, probs_b, Nv, v_b, MIDv, nullptr, o_b, MIDv, Nv, 64, Nv,
       nullptr, nullptr, 8, Bv * Hh, Hh,
       (long long)Hh * Nv * Nv, (long long)Nv * Nv, (long long)Nv * MIDv, 64,
       (long long)Nv * MIDv, 64);
  // x1 = O @ wo + bo + x
  gemm(stream, o_b, MIDv, wo_sa_b, Dv, x1, nullptr, Dv, rowsX, Dv, MIDv,
       sa_bo, x_in, 1 | 4, 1, 1, 0, 0, 0, 0, 0, 0);

  // ===================== Cross attention =====================
  adaln_kernel<<<rowsX, 256, 0, stream>>>(x1, ln1_w, ln1_b, g1, b1, act_b, Dv, Nv);
  adaln_kernel<<<rowsC, 256, 0, stream>>>(context, ca_lnc_w, ca_lnc_b, nullptr, nullptr,
                                          ctx_b, DCv, Mv);
  gemm(stream, act_b, Dv, wq_ca_b, MIDv, q_f, nullptr, MIDv, rowsX, MIDv, Dv,
       nullptr, nullptr, 0, 1, 1, 0, 0, 0, 0, 0, 0);
  rope_q_kernel<<<(rowsX * 512 + 255) / 256, 256, 0, stream>>>(q_f, q_b, (long long)rowsX * 512, Nv);
  gemm(stream, ctx_b, DCv, wkv_ca_b, 2 * MIDv, kv_f, nullptr, 2 * MIDv, rowsC, 2 * MIDv, DCv,
       nullptr, nullptr, 0, 1, 1, 0, 0, 0, 0, 0, 0);
  kv_proc_kernel<<<(rowsC * 512 + 255) / 256, 256, 0, stream>>>(kv_f, kT_b, v_b, Mv, (long long)rowsC * 512);
  // sim (M=1024, N=512, K=64, batch=32)
  gemm(stream, q_b, MIDv, kT_b, Mv, sim_f, nullptr, Mv, Nv, Mv, 64,
       nullptr, nullptr, 0, Bv * Hh, Hh,
       (long long)Nv * MIDv, 64, (long long)Hh * 64 * Mv, (long long)64 * Mv,
       (long long)Hh * Nv * Mv, (long long)Nv * Mv);
  softmax_kernel<<<Bv * Hh * Nv, 256, 0, stream>>>(sim_f, probs_b, Mv, iscale);
  gemm(stream, probs_b, Mv, v_b, MIDv, nullptr, o_b, MIDv, Nv, 64, Mv,
       nullptr, nullptr, 8, Bv * Hh, Hh,
       (long long)Hh * Nv * Mv, (long long)Nv * Mv, (long long)Mv * MIDv, 64,
       (long long)Nv * MIDv, 64);
  gemm(stream, o_b, MIDv, wo_ca_b, Dv, x2, nullptr, Dv, rowsX, Dv, MIDv,
       ca_bo, x1, 1 | 4, 1, 1, 0, 0, 0, 0, 0, 0);

  // ===================== FFN =====================
  adaln_kernel<<<rowsX, 256, 0, stream>>>(x2, ln2_w, ln2_b, g2, b2, act_b, Dv, Nv);
  // h = gelu(act @ ff1 + b1) -> bf16
  gemm(stream, act_b, Dv, ff1_b, FFv, nullptr, h_b, FFv, rowsX, FFv, Dv,
       ff_b1, nullptr, 1 | 2 | 8, 1, 1, 0, 0, 0, 0, 0, 0);
  // out = h @ ff2 + b2 + x2
  gemm(stream, h_b, FFv, ff2_b, Dv, (float*)d_out, nullptr, Dv, rowsX, Dv, FFv,
       ff_b2, x2, 1 | 4, 1, 1, 0, 0, 0, 0, 0, 0);
}
